// FineAMT_25967372271768
// MI455X (gfx1250) — matrix operand order repaired and checked by reading, not compiled, not run on hardware
//
#include <hip/hip_runtime.h>
#include <hip/hip_bf16.h>
#include <math.h>

typedef __attribute__((ext_vector_type(16))) _Float16 v16h;
typedef __attribute__((ext_vector_type(8)))  float    v8f;
typedef __attribute__((ext_vector_type(4)))  _Float16 h4;
typedef __attribute__((ext_vector_type(2)))  _Float16 h2;
typedef __attribute__((ext_vector_type(4)))  unsigned int u32x4;
typedef __attribute__((ext_vector_type(8)))  int i32x8;
typedef __attribute__((ext_vector_type(4)))  int i32x4;

#define B_SZ     4
#define L_SEQ    2048
#define NTOK     (B_SZ * L_SEQ)     // 8192
#define DMODEL   512
#define N_MELS   128
#define D2_DIM   384
#define D_STATE  64
#define D_CONVW  4
#define HEADDIM  64
#define NBLOCKS  6
#define D_INNER  1024
#define NHEADS   16
#define CONV_DIM 1152               // D_INNER + 2*D_STATE
#define D_INPROJ 2192               // 2*D_INNER + 2*D_STATE + NHEADS
#define N_PITCH  128

enum { EP_NONE = 0, EP_GELU = 1, EP_SIGMOID_MASK = 2 };

// ---------------------------------------------------------------------------
// TDM: issue a 2D tensor_load_to_lds (f32 elements).  D# layout per CDNA5 ISA
// ch8: group0 = {count, lds_addr, global_addr, type=2}, group1 = {data_size,
// tensor dims / tile dims / stride0}.  Reads past tensor_dim0 return zero,
// which gives us free N-edge zero padding.  This toolchain exposes the 6-arg
// builtin: (u32x4 g0, i32x8 g1, i32x4 g2, i32x4 g3, i32x8, i32 cpol).
// ---------------------------------------------------------------------------
__device__ __forceinline__ void tdm_load_2d_f32(
    unsigned int lds_off, const float* gptr,
    unsigned int tensor_d0, unsigned int tensor_d1,
    unsigned int tile_d0, unsigned int tile_d1,
    unsigned long long stride0_elems)
{
  unsigned long long ga = (unsigned long long)(uintptr_t)gptr;
  u32x4 g0;
  g0[0] = 1u;                                         // count=1, user desc
  g0[1] = lds_off;                                    // lds_addr (bytes)
  g0[2] = (unsigned int)(ga & 0xffffffffu);           // global_addr[31:0]
  g0[3] = (unsigned int)((ga >> 32) & 0x1ffffffu)     // global_addr[56:32]
        | (2u << 30);                                 // type = 2 ("image")
  i32x8 g1;
  g1[0] = (int)(2u << 16);                            // data_size = 4 bytes
  g1[1] = (int)((tensor_d0 & 0xffffu) << 16);         // dim0[15:0] @ bits 63:48
  g1[2] = (int)(((tensor_d0 >> 16) & 0xffffu) | ((tensor_d1 & 0xffffu) << 16));
  g1[3] = (int)(((tensor_d1 >> 16) & 0xffffu) | ((tile_d0 & 0xffffu) << 16));
  g1[4] = (int)(tile_d1 & 0xffffu);                   // tile_dim1, tile_dim2=0
  g1[5] = (int)(unsigned int)(stride0_elems & 0xffffffffu);
  g1[6] = (int)(unsigned int)((stride0_elems >> 32) & 0xffffu);
  g1[7] = 0;
  i32x4 gz  = {0, 0, 0, 0};
  i32x8 gz8 = {0, 0, 0, 0, 0, 0, 0, 0};
  __builtin_amdgcn_tensor_load_to_lds(g0, g1, gz, gz, gz8, 0);
}

// ---------------------------------------------------------------------------
// Tiled WMMA GEMM:  C[M x N] = epilogue(A[M x K](lda) * W[K x N] + bias)
// 128 threads = 4 waves; 64x64 tile; each wave computes a 16(M) x 64(N) strip.
// A tile: vectorized global_load_b128 -> cvt -> ds_store_b64.
// B tile: TDM async tensor load (f32 staging) -> LDS convert/transpose with
//         paired-k f16 packing so fragment reads remain ds_load_b128.
// M % 64 == 0, K % 32 == 0; N arbitrary (TDM OOB zero-fill + store guard).
// ---------------------------------------------------------------------------
template <int EPI>
__global__ __launch_bounds__(128) void wmma_gemm(
    const float* __restrict__ A, int lda,
    const float* __restrict__ W,
    const float* __restrict__ bias,
    float* __restrict__ C,
    int M, int N, int K,
    const int* __restrict__ rowmask)
{
  __shared__ _Float16 lsA[64 * 32];   // [m][k]
  __shared__ _Float16 lsB[64 * 32];   // [n][k] (transposed on fill)
  __shared__ float    stagB[32 * 64]; // raw f32 B tile from TDM: [k][n]

  const int tid   = threadIdx.x;
  const int lane  = tid & 31;
  const int wave  = tid >> 5;
  const int hlf   = lane >> 4;   // which 16-lane half
  const int l15   = lane & 15;
  const int tileN = blockIdx.x * 64;
  const int tileM = blockIdx.y * 64;

  const unsigned int stagB_off = (unsigned int)(uintptr_t)(void*)stagB;
  const unsigned int remN = (unsigned int)(N - tileN);   // > 0 by grid sizing

  v8f acc[4];
#pragma unroll
  for (int j = 0; j < 4; ++j) {
    v8f z = {0.f, 0.f, 0.f, 0.f, 0.f, 0.f, 0.f, 0.f};
    acc[j] = z;
  }

  for (int kk = 0; kk < K; kk += 32) {
    // --- issue async TDM load of the B tile (32 k-rows x 64 n) ------------
    if (tid == 0) {
      tdm_load_2d_f32(stagB_off, &W[(size_t)kk * N + tileN],
                      remN, 32u, 64u, 32u, (unsigned long long)N);
    }

    // --- A tile: 64 rows x 32 k, float4 loads, packed f16 b64 stores ------
#pragma unroll
    for (int it = 0; it < 4; ++it) {
      int idx = tid + it * 128;          // 0..511, 4 elements each
      int r  = idx >> 3;
      int kq = (idx & 7) << 2;
      const float4 f = *(const float4*)&A[(size_t)(tileM + r) * lda + (kk + kq)];
      h4 hv;
      hv[0] = (_Float16)f.x; hv[1] = (_Float16)f.y;
      hv[2] = (_Float16)f.z; hv[3] = (_Float16)f.w;
      *(h4*)&lsA[r * 32 + kq] = hv;
    }

    if (tid == 0) __builtin_amdgcn_s_wait_tensorcnt(0);
    __syncthreads();                     // stagB + lsA visible to all

    // --- convert/transpose staging f32 [k][n] -> lsB f16 [n][k] -----------
    // pack (k, k+1) per n into one 32-bit store so frag reads stay b128
#pragma unroll
    for (int it = 0; it < 2; ++it) {
      int idx = tid + it * 128;          // 0..255, 8 elements each
      int n0 = (idx & 15) << 2;
      int k  = (idx >> 4) << 1;
      float4 f0 = *(const float4*)&stagB[k * 64 + n0];
      float4 f1 = *(const float4*)&stagB[(k + 1) * 64 + n0];
      h2 p;
      p[0] = (_Float16)f0.x; p[1] = (_Float16)f1.x; *(h2*)&lsB[(n0 + 0) * 32 + k] = p;
      p[0] = (_Float16)f0.y; p[1] = (_Float16)f1.y; *(h2*)&lsB[(n0 + 1) * 32 + k] = p;
      p[0] = (_Float16)f0.z; p[1] = (_Float16)f1.z; *(h2*)&lsB[(n0 + 2) * 32 + k] = p;
      p[0] = (_Float16)f0.w; p[1] = (_Float16)f1.w; *(h2*)&lsB[(n0 + 3) * 32 + k] = p;
    }
    __syncthreads();                     // lsB ready

    // --- A fragment (16x32 f16 layout, ISA 7.12.2): row = wave*16 + l15 ---
    v16h afrag;
#pragma unroll
    for (int e = 0; e < 16; ++e) {
      int v = e >> 1, s = e & 1;
      int k = ((v >> 2) << 4) + (hlf << 3) + ((v & 3) << 1) + s;
      afrag[e] = lsA[(wave * 16 + l15) * 32 + k];
    }

#pragma unroll
    for (int j = 0; j < 4; ++j) {
      // B fragment (32x16): col = j*16 + l15, k = (v>>2)*16 + hlf*8 + 2*(v&3) + s
      v16h bfrag;
#pragma unroll
      for (int e = 0; e < 16; ++e) {
        int v = e >> 1, s = e & 1;
        int k = ((v >> 2) << 4) + (hlf << 3) + ((v & 3) << 1) + s;
        bfrag[e] = lsB[(j * 16 + l15) * 32 + k];
      }
      acc[j] = __builtin_amdgcn_wmma_f32_16x16x32_f16(
          false, afrag, false, bfrag, (short)0, acc[j], false, false);
    }
    __syncthreads();                     // protect tiles for next iteration
  }

  // Store D: m = tileM + wave*16 + hlf*8 + r, n = tileN + j*16 + l15
#pragma unroll
  for (int j = 0; j < 4; ++j) {
    int n = tileN + j * 16 + l15;
    if (n < N) {
      float bv = bias ? bias[n] : 0.f;
#pragma unroll
      for (int r = 0; r < 8; ++r) {
        int m = tileM + wave * 16 + hlf * 8 + r;
        float val = acc[j][r] + bv;
        if (EPI == EP_GELU) {
          val = 0.5f * val * (1.0f + erff(val * 0.70710678118f));
        } else if (EPI == EP_SIGMOID_MASK) {
          float sg = 1.f / (1.f + __expf(-val));
          val = (rowmask[m] == 1) ? sg : 0.f;
        }
        C[(size_t)m * N + n] = val;
      }
    }
  }
}

// ---------------------------------------------------------------------------
// Embedding combine + LayerNorm: x = LN(select(e1,e2,type) + type_emb + pos)
// ---------------------------------------------------------------------------
__global__ __launch_bounds__(256) void k_embed_combine(
    const float* __restrict__ e1, const float* __restrict__ e2,
    const int* __restrict__ type_ids,
    const float* __restrict__ type_emb, const float* __restrict__ pos_emb,
    const float* __restrict__ w, const float* __restrict__ b,
    float* __restrict__ out)
{
  int m = blockIdx.x, l = m % L_SEQ, t = type_ids[m];
  int tc = t < 0 ? 0 : t;
  __shared__ float red[256];
  float v[2];
#pragma unroll
  for (int i = 0; i < 2; ++i) {
    int d = threadIdx.x + i * 256;
    float e = (t == 0) ? e1[(size_t)m * DMODEL + d]
            : (t == 1) ? e2[(size_t)m * DMODEL + d] : 0.f;
    v[i] = e + type_emb[tc * DMODEL + d] + pos_emb[(size_t)l * DMODEL + d];
  }
  red[threadIdx.x] = v[0] + v[1];
  __syncthreads();
  for (int s = 128; s > 0; s >>= 1) {
    if (threadIdx.x < s) red[threadIdx.x] += red[threadIdx.x + s];
    __syncthreads();
  }
  float mean = red[0] * (1.f / DMODEL);
  __syncthreads();
  v[0] -= mean; v[1] -= mean;
  red[threadIdx.x] = v[0] * v[0] + v[1] * v[1];
  __syncthreads();
  for (int s = 128; s > 0; s >>= 1) {
    if (threadIdx.x < s) red[threadIdx.x] += red[threadIdx.x + s];
    __syncthreads();
  }
  float rs = rsqrtf(red[0] * (1.f / DMODEL) + 1e-5f);
#pragma unroll
  for (int i = 0; i < 2; ++i) {
    int d = threadIdx.x + i * 256;
    float o = v[i] * rs * w[d] + b[d];
    out[(size_t)m * DMODEL + d] = (t == -1) ? 0.f : o;
  }
}

// Plain LayerNorm over 512 dims
__global__ __launch_bounds__(256) void k_layernorm(
    const float* __restrict__ x, const float* __restrict__ w,
    const float* __restrict__ b, float* __restrict__ out)
{
  int m = blockIdx.x;
  __shared__ float red[256];
  float v0 = x[(size_t)m * DMODEL + threadIdx.x];
  float v1 = x[(size_t)m * DMODEL + 256 + threadIdx.x];
  red[threadIdx.x] = v0 + v1;
  __syncthreads();
  for (int s = 128; s > 0; s >>= 1) {
    if (threadIdx.x < s) red[threadIdx.x] += red[threadIdx.x + s];
    __syncthreads();
  }
  float mean = red[0] * (1.f / DMODEL);
  __syncthreads();
  v0 -= mean; v1 -= mean;
  red[threadIdx.x] = v0 * v0 + v1 * v1;
  __syncthreads();
  for (int s = 128; s > 0; s >>= 1) {
    if (threadIdx.x < s) red[threadIdx.x] += red[threadIdx.x + s];
    __syncthreads();
  }
  float rs = rsqrtf(red[0] * (1.f / DMODEL) + 1e-5f);
  out[(size_t)m * DMODEL + threadIdx.x]       = v0 * rs * w[threadIdx.x] + b[threadIdx.x];
  out[(size_t)m * DMODEL + 256 + threadIdx.x] = v1 * rs * w[256 + threadIdx.x] + b[256 + threadIdx.x];
}

// dt = softplus(zx[..., 2176 + h] + dt_bias[h])
__global__ void k_dt(const float* __restrict__ zx,
                     const float* __restrict__ dt_bias,
                     float* __restrict__ dt)
{
  int idx = blockIdx.x * blockDim.x + threadIdx.x;
  if (idx >= NTOK * NHEADS) return;
  int m = idx / NHEADS, h = idx % NHEADS;
  float x = zx[(size_t)m * D_INPROJ + D_INNER + CONV_DIM + h] + dt_bias[h];
  dt[idx] = (x > 20.f) ? x : log1pf(__expf(x));
}

// causal depthwise conv (width 4) + SiLU over the xBC slice of zxbcdt
__global__ void k_conv(const float* __restrict__ zx,
                       const float* __restrict__ cw,
                       const float* __restrict__ cb,
                       float* __restrict__ out)
{
  int idx = blockIdx.x * blockDim.x + threadIdx.x;
  if (idx >= NTOK * CONV_DIM) return;
  int m = idx / CONV_DIM, c = idx % CONV_DIM;
  int b = m / L_SEQ, l = m % L_SEQ;
  float acc = cb[c];
#pragma unroll
  for (int k = 0; k < D_CONVW; ++k) {
    int ls = l - (D_CONVW - 1) + k;
    if (ls >= 0)
      acc += cw[k * CONV_DIM + c] *
             zx[((size_t)b * L_SEQ + ls) * D_INPROJ + D_INNER + c];
  }
  out[idx] = acc / (1.f + __expf(-acc));
}

// ---------------------------------------------------------------------------
// Selective-state scan: one block per (batch, head) chain, 64 threads,
// thread p owns row p of the 64x64 state (64 f32 in registers).
// ---------------------------------------------------------------------------
__global__ __launch_bounds__(64) void k_scan(
    const float* __restrict__ xconv, const float* __restrict__ dtbuf,
    const float* __restrict__ A_log, const float* __restrict__ Dp,
    float* __restrict__ ybuf)
{
  int bh = blockIdx.x;
  int b = bh >> 4, h = bh & 15;
  int p = threadIdx.x;
  __shared__ float Bs[D_STATE], Cs[D_STATE];
  float Aneg = -__expf(A_log[h]);
  float Dv = Dp[h];
  float hstate[D_STATE];
#pragma unroll
  for (int n = 0; n < D_STATE; ++n) hstate[n] = 0.f;

  for (int l = 0; l < L_SEQ; ++l) {
    size_t m = (size_t)b * L_SEQ + l;
    const float* xc = xconv + m * CONV_DIM;
    Bs[p] = xc[D_INNER + p];
    Cs[p] = xc[D_INNER + D_STATE + p];
    __syncthreads();
    float dtv = dtbuf[m * NHEADS + h];
    float dA = __expf(dtv * Aneg);
    float xp = xc[h * HEADDIM + p];
    float coef = dtv * xp;
    float y = 0.f;
#pragma unroll
    for (int n = 0; n < D_STATE; ++n) {
      hstate[n] = dA * hstate[n] + coef * Bs[n];
      y += hstate[n] * Cs[n];
    }
    ybuf[m * D_INNER + h * HEADDIM + p] = y + Dv * xp;
    __syncthreads();
  }
}

// y = y * silu(z); RMSNorm over 1024 dims * norm_w
__global__ __launch_bounds__(256) void k_gated_rmsnorm(
    const float* __restrict__ zx, const float* __restrict__ ybuf,
    const float* __restrict__ nw, float* __restrict__ out)
{
  int m = blockIdx.x;
  __shared__ float red[256];
  float v[4];
  float ss = 0.f;
#pragma unroll
  for (int i = 0; i < 4; ++i) {
    int d = threadIdx.x + i * 256;
    float z = zx[(size_t)m * D_INPROJ + d];
    float y = ybuf[(size_t)m * D_INNER + d];
    float sv = z / (1.f + __expf(-z));
    v[i] = y * sv;
    ss += v[i] * v[i];
  }
  red[threadIdx.x] = ss;
  __syncthreads();
  for (int s = 128; s > 0; s >>= 1) {
    if (threadIdx.x < s) red[threadIdx.x] += red[threadIdx.x + s];
    __syncthreads();
  }
  float scale = rsqrtf(red[0] * (1.f / D_INNER) + 1e-5f);
#pragma unroll
  for (int i = 0; i < 4; ++i) {
    int d = threadIdx.x + i * 256;
    out[(size_t)m * D_INNER + d] = v[i] * scale * nw[d];
  }
}

// ---------------------------------------------------------------------------
// Host-side orchestration (graph-capture safe: launches only)
// ---------------------------------------------------------------------------
extern "C" void kernel_launch(void* const* d_in, const int* in_sizes, int n_in,
                              void* d_out, int out_size, void* d_ws, size_t ws_size,
                              hipStream_t stream) {
  const float* seq      = (const float*)d_in[0];
  const int*   type_ids = (const int*)d_in[1];
  const float* w1a = (const float*)d_in[2];
  const float* b1a = (const float*)d_in[3];
  const float* w1b = (const float*)d_in[4];
  const float* b1b = (const float*)d_in[5];
  const float* w2a = (const float*)d_in[6];
  const float* b2a = (const float*)d_in[7];
  const float* w2b = (const float*)d_in[8];
  const float* b2b = (const float*)d_in[9];
  const float* type_emb = (const float*)d_in[10];
  const float* pos_emb  = (const float*)d_in[11];
  const float* emb_ln_w = (const float*)d_in[12];
  const float* emb_ln_b = (const float*)d_in[13];
  const float* in_proj  = (const float*)d_in[14];
  const float* conv_w   = (const float*)d_in[15];
  const float* conv_b   = (const float*)d_in[16];
  const float* dt_bias  = (const float*)d_in[17];
  const float* A_log    = (const float*)d_in[18];
  const float* Dp       = (const float*)d_in[19];
  const float* mnorm_w  = (const float*)d_in[20];
  const float* out_proj = (const float*)d_in[21];
  const float* fin_ln_w = (const float*)d_in[22];
  const float* fin_ln_b = (const float*)d_in[23];
  const float* on_w  = (const float*)d_in[24];
  const float* on_b  = (const float*)d_in[25];
  const float* off_w = (const float*)d_in[26];
  const float* off_b = (const float*)d_in[27];
  const float* fr_w  = (const float*)d_in[28];
  const float* fr_b  = (const float*)d_in[29];

  // workspace arena
  char* ws = (char*)d_ws;
  const size_t SZ_X  = (size_t)NTOK * DMODEL   * 4;
  const size_t SZ_ZX = (size_t)NTOK * D_INPROJ * 4;
  const size_t SZ_XC = (size_t)NTOK * CONV_DIM * 4;
  const size_t SZ_DT = (size_t)NTOK * NHEADS   * 4;
  const size_t SZ_Y  = (size_t)NTOK * D_INNER  * 4;
  float* X  = (float*)(ws);
  float* ZX = (float*)(ws + SZ_X);
  float* XC = (float*)(ws + SZ_X + SZ_ZX);
  float* DT = (float*)(ws + SZ_X + SZ_ZX + SZ_XC);
  float* Y  = (float*)(ws + SZ_X + SZ_ZX + SZ_XC + SZ_DT);
  float* YN = (float*)(ws + SZ_X + SZ_ZX + SZ_XC + SZ_DT + SZ_Y);
  float* T1 = (float*)(ws + SZ_X + SZ_ZX + SZ_XC + SZ_DT + 2 * SZ_Y);
  float* T2 = (float*)(ws + SZ_X + SZ_ZX + SZ_XC + SZ_DT + 2 * SZ_Y + SZ_X);
  float* E1 = (float*)(ws + SZ_X + SZ_ZX + SZ_XC + SZ_DT + 2 * SZ_Y + 2 * SZ_X);

  const int MT = NTOK / 64;   // 128 M-tiles
  dim3 blk(128);

  // ---- embedding MLPs ----
  wmma_gemm<EP_GELU><<<dim3(DMODEL / 64, MT), blk, 0, stream>>>(
      seq, D2_DIM, w1a, b1a, T1, NTOK, DMODEL, N_MELS, nullptr);
  wmma_gemm<EP_NONE><<<dim3(DMODEL / 64, MT), blk, 0, stream>>>(
      T1, DMODEL, w1b, b1b, E1, NTOK, DMODEL, DMODEL, nullptr);
  wmma_gemm<EP_GELU><<<dim3(DMODEL / 64, MT), blk, 0, stream>>>(
      seq, D2_DIM, w2a, b2a, T1, NTOK, DMODEL, D2_DIM, nullptr);
  wmma_gemm<EP_NONE><<<dim3(DMODEL / 64, MT), blk, 0, stream>>>(
      T1, DMODEL, w2b, b2b, T2, NTOK, DMODEL, DMODEL, nullptr);
  k_embed_combine<<<NTOK, 256, 0, stream>>>(E1, T2, type_ids, type_emb, pos_emb,
                                            emb_ln_w, emb_ln_b, X);

  // ---- 6 Mamba2 blocks ----
  for (int i = 0; i < NBLOCKS; ++i) {
    wmma_gemm<EP_NONE><<<dim3((D_INPROJ + 63) / 64, MT), blk, 0, stream>>>(
        X, DMODEL, in_proj + (size_t)i * DMODEL * D_INPROJ, nullptr,
        ZX, NTOK, D_INPROJ, DMODEL, nullptr);
    k_dt<<<(NTOK * NHEADS + 255) / 256, 256, 0, stream>>>(ZX, dt_bias + i * NHEADS, DT);
    k_conv<<<(NTOK * CONV_DIM + 255) / 256, 256, 0, stream>>>(
        ZX, conv_w + (size_t)i * D_CONVW * CONV_DIM, conv_b + (size_t)i * CONV_DIM, XC);
    k_scan<<<B_SZ * NHEADS, 64, 0, stream>>>(XC, DT, A_log + i * NHEADS, Dp + i * NHEADS, Y);
    k_gated_rmsnorm<<<NTOK, 256, 0, stream>>>(ZX, Y, mnorm_w + (size_t)i * D_INNER, YN);
    wmma_gemm<EP_NONE><<<dim3(DMODEL / 64, MT), blk, 0, stream>>>(
        YN, D_INNER, out_proj + (size_t)i * D_INNER * DMODEL, nullptr,
        X, NTOK, DMODEL, D_INNER, nullptr);
  }

  // ---- final LN + heads ----
  k_layernorm<<<NTOK, 256, 0, stream>>>(X, fin_ln_w, fin_ln_b, T1);
  float* out = (float*)d_out;
  wmma_gemm<EP_SIGMOID_MASK><<<dim3(N_PITCH / 64, MT), blk, 0, stream>>>(
      T1, DMODEL, on_w, on_b, out, NTOK, N_PITCH, DMODEL, type_ids);
  wmma_gemm<EP_SIGMOID_MASK><<<dim3(N_PITCH / 64, MT), blk, 0, stream>>>(
      T1, DMODEL, off_w, off_b, out + (size_t)NTOK * N_PITCH, NTOK, N_PITCH, DMODEL, type_ids);
  wmma_gemm<EP_SIGMOID_MASK><<<dim3(N_PITCH / 64, MT), blk, 0, stream>>>(
      T1, DMODEL, fr_w, fr_b, out + 2 * (size_t)NTOK * N_PITCH, NTOK, N_PITCH, DMODEL, type_ids);
}
